// BlocksCore_58153857188094
// MI455X (gfx1250) — matrix operand order repaired and checked by reading, not compiled, not run on hardware
//
#include <hip/hip_runtime.h>
#include <hip/hip_bf16.h>
#include <cstddef>

// ---------- types ----------
typedef __bf16 bf16;
typedef __attribute__((ext_vector_type(16))) __bf16 v16bf;
typedef __attribute__((ext_vector_type(8)))  __bf16 bf16x8;
typedef __attribute__((ext_vector_type(8)))  float  v8f;
typedef int v4i __attribute__((__vector_size__(16)));  // matches builtin's param type

#define LDS_STRIDE 40   // padded row stride in halves (80B -> conflict-free frag reads)

#if defined(__has_builtin)
#if __has_builtin(__builtin_amdgcn_global_load_async_to_lds_b128)
#define USE_ASYNC_LDS 1
#endif
#endif

__device__ __forceinline__ float sigmoidf_(float x) { return 1.0f / (1.0f + __expf(-x)); }

#ifdef USE_ASYNC_LDS
#define GLOBAL_AS __attribute__((address_space(1)))
#define LDS_AS    __attribute__((address_space(3)))
__device__ __forceinline__ void async_ld16(const bf16* g, bf16* l) {
    __builtin_amdgcn_global_load_async_to_lds_b128(
        (GLOBAL_AS v4i*)(GLOBAL_AS void*)(g),
        (LDS_AS v4i*)(LDS_AS void*)(l), 0, 0);
}
#endif

// ======================================================================
// Generic WMMA GEMM:  C[m,n] = sum_k A[m,k] * B[n,k]   (A: MxK, B: NxK, both
// bf16 row-major over K; C fp32). 128x128 tile / workgroup, wave tile 32x64,
// K-step 32, double-buffered LDS, one barrier per K-step.
// Requires M%128==0, N%128==0, K%32==0, 16B-aligned A/B, lda/ldb%8==0.
// ======================================================================
__global__ __launch_bounds__(256)
void gemm128_bf16_nt(const bf16* __restrict__ A, int lda,
                     const bf16* __restrict__ B, int ldb,
                     float* __restrict__ C, int ldc, int K)
{
    __shared__ bf16 As[2][128 * LDS_STRIDE];
    __shared__ bf16 Bs[2][128 * LDS_STRIDE];

    const int tid   = threadIdx.x;
    const int lane  = tid & 31;
    const int wave  = tid >> 5;
    const int waveM = wave & 3;   // 4 waves along M
    const int waveN = wave >> 2;  // 2 waves along N
    const int m0 = blockIdx.y * 128;
    const int n0 = blockIdx.x * 128;
    const int l15 = lane & 15;
    const int hi  = lane >> 4;    // 0/1 half-wave

    // Staging: each thread moves 4 x 16B per tile (A row lr & lr+64, B likewise)
    const int    lr    = tid >> 2;          // 0..63
    const int    ls    = (tid & 3) * 8;     // 0,8,16,24 halves
    const size_t aoff  = (size_t)(m0 + lr) * lda + ls;
    const size_t boff  = (size_t)(n0 + lr) * ldb + ls;
    const size_t astep = (size_t)64 * lda;
    const size_t bstep = (size_t)64 * ldb;
    const int    loffA = lr * LDS_STRIDE + ls;
    const int    loffB = loffA + 64 * LDS_STRIDE;

    // Fragment read bases (documented CDNA5 A 16x32 / B 32x16 VGPR layouts)
    const int aread = (waveM * 32 + l15) * LDS_STRIDE + hi * 8;
    const int bread = (waveN * 64 + l15) * LDS_STRIDE + hi * 8;

    v8f zero = {};
    v8f acc[2][4];
#pragma unroll
    for (int t = 0; t < 2; ++t)
#pragma unroll
        for (int j = 0; j < 4; ++j) acc[t][j] = zero;

    union Frag { v16bf v; bf16x8 h[2]; };

#ifdef USE_ASYNC_LDS
    // ---------------- async-DMA double-buffered pipeline ----------------
    async_ld16(A + aoff,         &As[0][loffA]);
    async_ld16(A + aoff + astep, &As[0][loffB]);
    async_ld16(B + boff,         &Bs[0][loffA]);
    async_ld16(B + boff + bstep, &Bs[0][loffB]);
    int cur = 0;
    for (int k0 = 0; k0 < K; k0 += 32, cur ^= 1) {
        asm volatile("s_wait_asynccnt 0" ::: "memory");  // this wave's tile landed
        __syncthreads();                                 // everyone's tile landed
        if (k0 + 32 < K) {
            const int nxt = cur ^ 1;
            const size_t kn = (size_t)(k0 + 32);
            async_ld16(A + aoff + kn,         &As[nxt][loffA]);
            async_ld16(A + aoff + astep + kn, &As[nxt][loffB]);
            async_ld16(B + boff + kn,         &Bs[nxt][loffA]);
            async_ld16(B + boff + bstep + kn, &Bs[nxt][loffB]);
        }
        Frag af[2], bf[4];
#pragma unroll
        for (int t = 0; t < 2; ++t) {
            const bf16* p = &As[cur][aread + t * 16 * LDS_STRIDE];
            af[t].h[0] = *(const bf16x8*)(p);
            af[t].h[1] = *(const bf16x8*)(p + 16);
        }
#pragma unroll
        for (int j = 0; j < 4; ++j) {
            const bf16* p = &Bs[cur][bread + j * 16 * LDS_STRIDE];
            bf[j].h[0] = *(const bf16x8*)(p);
            bf[j].h[1] = *(const bf16x8*)(p + 16);
        }
#pragma unroll
        for (int t = 0; t < 2; ++t)
#pragma unroll
            for (int j = 0; j < 4; ++j)
                acc[t][j] = __builtin_amdgcn_wmma_f32_16x16x32_bf16(
                    false, af[t].v, false, bf[j].v, (short)0, acc[t][j], false, false);
    }
#else
    // ---------------- register double-buffered pipeline ----------------
    bf16x8 ra0 = *(const bf16x8*)&A[aoff];
    bf16x8 ra1 = *(const bf16x8*)&A[aoff + astep];
    bf16x8 rb0 = *(const bf16x8*)&B[boff];
    bf16x8 rb1 = *(const bf16x8*)&B[boff + bstep];
    *(bf16x8*)&As[0][loffA] = ra0;
    *(bf16x8*)&As[0][loffB] = ra1;
    *(bf16x8*)&Bs[0][loffA] = rb0;
    *(bf16x8*)&Bs[0][loffB] = rb1;
    int cur = 0;
    for (int k0 = 0; k0 < K; k0 += 32, cur ^= 1) {
        const bool nx = (k0 + 32 < K);
        if (nx) {
            const size_t kn = (size_t)(k0 + 32);
            ra0 = *(const bf16x8*)&A[aoff + kn];
            ra1 = *(const bf16x8*)&A[aoff + astep + kn];
            rb0 = *(const bf16x8*)&B[boff + kn];
            rb1 = *(const bf16x8*)&B[boff + bstep + kn];
            __builtin_prefetch(&A[aoff + kn + 32], 0, 1);          // global_prefetch_b8
            __builtin_prefetch(&B[boff + kn + 32], 0, 1);
        }
        __syncthreads();
        Frag af[2], bf[4];
#pragma unroll
        for (int t = 0; t < 2; ++t) {
            const bf16* p = &As[cur][aread + t * 16 * LDS_STRIDE];
            af[t].h[0] = *(const bf16x8*)(p);
            af[t].h[1] = *(const bf16x8*)(p + 16);
        }
#pragma unroll
        for (int j = 0; j < 4; ++j) {
            const bf16* p = &Bs[cur][bread + j * 16 * LDS_STRIDE];
            bf[j].h[0] = *(const bf16x8*)(p);
            bf[j].h[1] = *(const bf16x8*)(p + 16);
        }
#pragma unroll
        for (int t = 0; t < 2; ++t)
#pragma unroll
            for (int j = 0; j < 4; ++j)
                acc[t][j] = __builtin_amdgcn_wmma_f32_16x16x32_bf16(
                    false, af[t].v, false, bf[j].v, (short)0, acc[t][j], false, false);
        if (nx) {
            const int nxt = cur ^ 1;
            *(bf16x8*)&As[nxt][loffA] = ra0;
            *(bf16x8*)&As[nxt][loffB] = ra1;
            *(bf16x8*)&Bs[nxt][loffA] = rb0;
            *(bf16x8*)&Bs[nxt][loffB] = rb1;
        }
    }
#endif

    // Epilogue: C layout per 16x16 tile: lane l15 = col; rows = hi*8 + r
#pragma unroll
    for (int t = 0; t < 2; ++t)
#pragma unroll
        for (int j = 0; j < 4; ++j) {
            const int col   = n0 + waveN * 64 + j * 16 + l15;
            const int rbase = m0 + waveM * 32 + t * 16 + hi * 8;
#pragma unroll
            for (int r = 0; r < 8; ++r)
                C[(size_t)(rbase + r) * ldc + col] = acc[t][j][r];
        }
}

// ======================================================================
// Conversion / packing kernels
// ======================================================================
__global__ void cvt_hx_to_acat(const float* __restrict__ hx, bf16* __restrict__ Acat, int total) {
    int idx = blockIdx.x * blockDim.x + threadIdx.x;
    if (idx >= total) return;
    int b = idx >> 11, k = idx & 2047;
    Acat[(size_t)b * 4096 + 2048 + k] = (bf16)hx[idx];
}

__global__ void cvt_f32_bf16(const float* __restrict__ src, bf16* __restrict__ dst, int total) {
    int idx = blockIdx.x * blockDim.x + threadIdx.x;
    if (idx < total) dst[idx] = (bf16)src[idx];
}

// WkvT (384 x 1024): rows 0..63 = wk_inp[1,:,:]^T ; 64..319 = wv_inp[1,:,:]^T ; 320..383 = 0
__global__ void build_wkvT(const float* __restrict__ wk, const float* __restrict__ wv,
                           bf16* __restrict__ dst) {
    int idx = blockIdx.x * blockDim.x + threadIdx.x;
    if (idx >= 384 * 1024) return;
    int d = idx >> 10, k = idx & 1023;
    float v = 0.0f;
    if (d < 64)       v = wk[65536 + k * 64 + d];          // wk_inp[1,k,d]
    else if (d < 320) v = wv[262144 + k * 256 + (d - 64)]; // wv_inp[1,k,d-64]
    dst[idx] = (bf16)v;
}

// Wq1T (512 x 2048): block-diagonal transpose of wq_inp (8,256,64)
__global__ void build_wq1T(const float* __restrict__ wq, bf16* __restrict__ dst) {
    int idx = blockIdx.x * blockDim.x + threadIdx.x;
    if (idx >= 512 * 2048) return;
    int r = idx >> 11, k = idx & 2047;
    int q = r >> 6, d = r & 63;
    int kb = k - q * 256;
    float v = (kb >= 0 && kb < 256) ? wq[(size_t)q * 256 * 64 + kb * 64 + d] : 0.0f;
    dst[idx] = (bf16)v;
}

// Wcat (8192 x 4096) = [w_ih | w_hh] in bf16
__global__ void build_wcat(const float* __restrict__ w_ih, const float* __restrict__ w_hh,
                           bf16* __restrict__ dst, int total) {
    int idx = blockIdx.x * blockDim.x + threadIdx.x;
    if (idx >= total) return;
    int row = idx >> 12, col = idx & 4095;
    float v = (col < 2048) ? w_ih[(size_t)row * 2048 + col]
                           : w_hh[(size_t)row * 2048 + (col - 2048)];
    dst[idx] = (bf16)v;
}

// Transpose+convert: dst[m*dstMatStride + d*K + k] = src[(m*K + k)*D + d]
__global__ void tcvt_kernel(const float* __restrict__ src, bf16* __restrict__ dst,
                            int nmat, int K, int D, int dstMatStride) {
    int idx = blockIdx.x * blockDim.x + threadIdx.x;
    int tot = nmat * K * D;
    if (idx >= tot) return;
    int m = idx / (K * D), rem = idx % (K * D);
    int k = rem / D, d = rem % D;
    dst[(size_t)m * dstMatStride + (size_t)d * K + k] = (bf16)src[idx];
}

// ======================================================================
// Attention-1 epilogue: scores, collapsed softmax, top-k mask, inp_flat->Acat
// ======================================================================
__global__ __launch_bounds__(256)
void att1_kernel(const float* __restrict__ qw,    // 2048 x 512
                 const float* __restrict__ k1v1,  // 2048 x 384 (k1 | v1 | pad)
                 bf16* __restrict__ Acat,         // 2048 x 4096 (writes cols 0..2047)
                 float* __restrict__ maskblk)     // 2048 x 8
{
    const int b = blockIdx.x, tid = threadIdx.x;
    __shared__ float k1s[64];
    __shared__ float sred[8];
    __shared__ float sigs[8];
    if (tid < 64) k1s[tid] = k1v1[(size_t)b * 384 + tid];
    __syncthreads();

    const int q = tid >> 5, ln = tid & 31;
    float p = qw[(size_t)b * 512 + q * 64 + ln] * k1s[ln] +
              qw[(size_t)b * 512 + q * 64 + 32 + ln] * k1s[32 + ln];
#pragma unroll
    for (int off = 16; off > 0; off >>= 1) p += __shfl_down(p, off);
    if (ln == 0) sred[q] = p * 0.125f;  // / sqrt(64)
    __syncthreads();

    if (tid < 8) {
        float sq = sred[tid];
        int cnt = 0;
#pragma unroll
        for (int pt = 0; pt < 8; ++pt) {
            float sp = sred[pt];
            cnt += (sp < sq || (sp == sq && pt < tid)) ? 1 : 0;
        }
        float keep = (cnt >= 4) ? 1.0f : 0.0f;  // drop the 4 smallest scores
        maskblk[(size_t)b * 8 + tid] = keep;
        sigs[tid] = sigmoidf_(sq);              // attention weight on the real key
    }
    __syncthreads();

    const float v1 = k1v1[(size_t)b * 384 + 64 + tid];  // tid = jj in 0..255
#pragma unroll
    for (int qq = 0; qq < 8; ++qq)
        Acat[(size_t)b * 4096 + qq * 256 + tid] = (bf16)(sigs[qq] * v1);
}

// ======================================================================
// LSTM pointwise
// ======================================================================
__global__ void lstm_kernel(const float* __restrict__ gates, const float* __restrict__ b_ih,
                            const float* __restrict__ b_hh, const float* __restrict__ cx,
                            float* __restrict__ cnew, float* __restrict__ hnew,
                            bf16* __restrict__ hbf, int total) {
    int idx = blockIdx.x * blockDim.x + threadIdx.x;
    if (idx >= total) return;
    int b = idx >> 11, j = idx & 2047;
    const float* g = gates + (size_t)b * 8192;
    float gi = g[j]        + b_ih[j]        + b_hh[j];
    float gf = g[2048 + j] + b_ih[2048 + j] + b_hh[2048 + j];
    float gg = g[4096 + j] + b_ih[4096 + j] + b_hh[4096 + j];
    float go = g[6144 + j] + b_ih[6144 + j] + b_hh[6144 + j];
    float c = sigmoidf_(gf) * cx[idx] + sigmoidf_(gi) * tanhf(gg);
    float h = sigmoidf_(go) * tanhf(c);
    cnew[idx] = c;
    hnew[idx] = h;
    hbf[idx]  = (bf16)h;
}

// ======================================================================
// Block self-attention (4 heads, 8 keys): qkv -> c_out (bf16)
// ======================================================================
__global__ __launch_bounds__(256)
void att2_kernel(const float* __restrict__ qkv,  // 2048 x 3072  (8 blocks x [q128|k128|v128])
                 bf16* __restrict__ coutbf)      // (2048*8) x 128
{
    const int b = blockIdx.x, tid = threadIdx.x;
    __shared__ float qs[8][128], ks[8][128], vs[8][128];
    __shared__ float sc[4][8][8];
    const float* base = qkv + (size_t)b * 3072;
    for (int i = tid; i < 1024; i += 256) {
        int n = i >> 7, d = i & 127;
        qs[n][d] = base[n * 384 + d];
        ks[n][d] = base[n * 384 + 128 + d];
        vs[n][d] = base[n * 384 + 256 + d];
    }
    __syncthreads();

    const int h = tid >> 6, qi = (tid >> 3) & 7, kj = tid & 7;
    float s = 0.0f;
#pragma unroll
    for (int d = 0; d < 32; ++d) s += qs[qi][h * 32 + d] * ks[kj][h * 32 + d];
    sc[h][qi][kj] = s * 0.17677669529663689f;  // 1/sqrt(32)
    __syncthreads();

    for (int o = tid; o < 1024; o += 256) {
        int i2 = o >> 7, c = o & 127, h2 = c >> 5;
        float mx = sc[h2][i2][0];
#pragma unroll
        for (int j = 1; j < 8; ++j) mx = fmaxf(mx, sc[h2][i2][j]);
        float den = 0.0f, num = 0.0f;
#pragma unroll
        for (int j = 0; j < 8; ++j) {
            float e = __expf(sc[h2][i2][j] - mx);
            den += e;
            num += e * vs[j][c];
        }
        coutbf[(size_t)b * 1024 + o] = (bf16)(num / den);
    }
}

// ======================================================================
// Final: fc/gate bias+activations, residual, mask merge -> d_out
// ======================================================================
__global__ void final_kernel(const float* __restrict__ fgC,  // 16384 x 512 ([fc|gate])
                             const float* __restrict__ fc_b, const float* __restrict__ gate_b,
                             const float* __restrict__ hnew, const float* __restrict__ cnew,
                             const float* __restrict__ hx, const float* __restrict__ cx,
                             const float* __restrict__ maskblk,
                             float* __restrict__ out, int total) {
    int idx = blockIdx.x * blockDim.x + threadIdx.x;
    if (idx >= total) return;
    int b = idx >> 11, j = idx & 2047;
    int n = j >> 8, jj = j & 255;
    const float* row = fgC + ((size_t)b * 8 + n) * 512;
    float o  = row[jj] + fc_b[jj];
    float gt = sigmoidf_(row[256 + jj] + gate_b[jj]);
    float hatt = gt * tanhf(o);
    float hn2 = hnew[idx] + hatt;
    float m = maskblk[(size_t)b * 8 + n];
    out[idx]             = m * hn2       + (1.0f - m) * hx[idx];
    out[total + idx]     = m * cnew[idx] + (1.0f - m) * cx[idx];
    out[2 * total + idx] = m;
}

// ======================================================================
// Host: layout workspace + launch pipeline
// ======================================================================
extern "C" void kernel_launch(void* const* d_in, const int* in_sizes, int n_in,
                              void* d_out, int out_size, void* d_ws, size_t ws_size,
                              hipStream_t stream) {
    const float* inp    = (const float*)d_in[0];
    const float* hx     = (const float*)d_in[1];
    const float* cx     = (const float*)d_in[2];
    const float* wq_inp = (const float*)d_in[3];
    const float* wk_inp = (const float*)d_in[4];
    const float* wv_inp = (const float*)d_in[5];
    const float* w_ih   = (const float*)d_in[6];
    const float* w_hh   = (const float*)d_in[7];
    const float* b_ih   = (const float*)d_in[8];
    const float* b_hh   = (const float*)d_in[9];
    const float* wq_c   = (const float*)d_in[10];
    const float* wk_c   = (const float*)d_in[11];
    const float* wv_c   = (const float*)d_in[12];
    const float* fc_w   = (const float*)d_in[13];
    const float* fc_b   = (const float*)d_in[14];
    const float* gate_w = (const float*)d_in[15];
    const float* gate_b = (const float*)d_in[16];
    float* out = (float*)d_out;

    char* ws = (char*)d_ws;
    size_t off = 0;
    auto take = [&](size_t bytes) -> char* {
        char* p = ws + off;
        off = (off + bytes + 255) & ~(size_t)255;
        return p;
    };
    bf16*  Acat   = (bf16*)take(2048ull * 4096 * 2);   // [inp_flat | hx] bf16
    bf16*  Wcat   = (bf16*)take(8192ull * 4096 * 2);   // [w_ih | w_hh] bf16
    char*  region = take(67108864);                    // gates, later qkv/cout/fg
    float* gates  = (float*)region;                    // 2048 x 8192 f32
    float* qkvbuf = (float*)region;                    // 2048 x 3072 f32 (after gates dead)
    bf16*  coutbf = (bf16*)(region + 25165824);        // 16384 x 128 bf16
    float* fgC    = (float*)(region + 25165824 + 4194304);  // 16384 x 512 f32
    bf16*  inpbf  = (bf16*)take(2048ull * 1024 * 2);
    bf16*  WkvT   = (bf16*)take(384ull * 1024 * 2);
    bf16*  Wq1T   = (bf16*)take(512ull * 2048 * 2);
    float* k1v1   = (float*)take(2048ull * 384 * 4);
    float* qw     = (float*)take(2048ull * 512 * 4);
    float* maskbk = (float*)take(2048ull * 8 * 4);
    float* cnew   = (float*)take(2048ull * 2048 * 4);
    float* hnew   = (float*)take(2048ull * 2048 * 4);
    bf16*  hbf    = (bf16*)take(2048ull * 2048 * 2);
    bf16*  WqkvT  = (bf16*)take(8ull * 384 * 256 * 2); // per block: [q|k|v]^T, 384 x 256
    bf16*  FGt    = (bf16*)take(512ull * 128 * 2);     // [fc^T | gate^T]
    (void)ws_size; (void)in_sizes; (void)n_in; (void)out_size;

    const int T = 256;
    auto g1 = [&](int n) { return dim3((n + T - 1) / T); };

    // ---- pack / convert ----
    cvt_hx_to_acat<<<g1(2048 * 2048), T, 0, stream>>>(hx, Acat, 2048 * 2048);
    cvt_f32_bf16<<<g1(2048 * 1024), T, 0, stream>>>(inp, inpbf, 2048 * 1024);
    build_wkvT<<<g1(384 * 1024), T, 0, stream>>>(wk_inp, wv_inp, WkvT);
    build_wq1T<<<g1(512 * 2048), T, 0, stream>>>(wq_inp, Wq1T);
    build_wcat<<<g1(8192 * 4096), T, 0, stream>>>(w_ih, w_hh, Wcat, 8192 * 4096);
    tcvt_kernel<<<g1(8 * 256 * 128), T, 0, stream>>>(wq_c, WqkvT,             8, 256, 128, 384 * 256);
    tcvt_kernel<<<g1(8 * 256 * 128), T, 0, stream>>>(wk_c, WqkvT + 128 * 256, 8, 256, 128, 384 * 256);
    tcvt_kernel<<<g1(8 * 256 * 128), T, 0, stream>>>(wv_c, WqkvT + 256 * 256, 8, 256, 128, 384 * 256);
    tcvt_kernel<<<g1(128 * 256), T, 0, stream>>>(fc_w,   FGt,             1, 128, 256, 256 * 128);
    tcvt_kernel<<<g1(128 * 256), T, 0, stream>>>(gate_w, FGt + 256 * 128, 1, 128, 256, 256 * 128);

    // ---- attention-1 projections (WMMA) ----
    gemm128_bf16_nt<<<dim3(3, 16), T, 0, stream>>>(inpbf, 1024, WkvT, 1024, k1v1, 384, 1024);
    gemm128_bf16_nt<<<dim3(4, 16), T, 0, stream>>>(Acat + 2048, 4096, Wq1T, 2048, qw, 512, 2048);
    att1_kernel<<<dim3(2048), T, 0, stream>>>(qw, k1v1, Acat, maskbk);

    // ---- the big LSTM GEMM: gates = [inp_flat|hx] @ [w_ih|w_hh]^T  (WMMA) ----
    gemm128_bf16_nt<<<dim3(64, 16), T, 0, stream>>>(Acat, 4096, Wcat, 4096, gates, 8192, 4096);
    lstm_kernel<<<g1(2048 * 2048), T, 0, stream>>>(gates, b_ih, b_hh, cx, cnew, hnew, hbf,
                                                   2048 * 2048);

    // ---- block self-attention q/k/v projections (WMMA, per block) ----
    for (int n = 0; n < 8; ++n) {
        gemm128_bf16_nt<<<dim3(3, 16), T, 0, stream>>>(
            hbf + n * 256, 2048, WqkvT + (size_t)n * 384 * 256, 256,
            qkvbuf + n * 384, 3072, 256);
    }
    att2_kernel<<<dim3(2048), T, 0, stream>>>(qkvbuf, coutbf);

    // ---- fc & gate projections (WMMA, fused) ----
    gemm128_bf16_nt<<<dim3(4, 128), T, 0, stream>>>(coutbf, 128, FGt, 128, fgC, 512, 128);

    // ---- final merge ----
    final_kernel<<<g1(2048 * 2048), T, 0, stream>>>(fgC, fc_b, gate_b, hnew, cnew, hx, cx,
                                                    maskbk, out, 2048 * 2048);
}
